// TransformerEncoder_34763465294525
// MI455X (gfx1250) — compile-verified, matrix operand order checked by reading, not run on hardware
//
#include <hip/hip_runtime.h>
#include <stdint.h>

// ---------------------------------------------------------------------------
// Transformer encoder (L=3, B=2, S=2048, D=1024, H=16, Dh=64, DF=4096) on
// MI455X / gfx1250. All GEMM math goes through v_wmma_f32_16x16x32_bf16
// (f32 data converted to bf16 via native v_cvt, f32 accumulation).
// Requires ~184 MB of d_ws scratch.
// ---------------------------------------------------------------------------

constexpr int Bc = 2, Sc = 2048, Dc = 1024, Hc = 16, Dhc = 64, DFc = 4096, Lc = 3;

typedef __bf16 bf16;
typedef __attribute__((ext_vector_type(16))) __bf16 bf16x16;
typedef __attribute__((ext_vector_type(8)))  __bf16 bf16x8;
typedef __attribute__((ext_vector_type(4)))  __bf16 bf16x4;
typedef __attribute__((ext_vector_type(8)))  float  f32x8;
typedef __attribute__((ext_vector_type(4)))  float  f32x4;

#define DEV __device__ __forceinline__

DEV bf16 f2bf(float f) { return static_cast<bf16>(f); }  // native v_cvt (RNE)

DEV void store4bf(bf16* p, f32x4 v) {  // 8-byte packed LDS/global store
  *(bf16x4*)p = __builtin_convertvector(v, bf16x4);
}

DEV f32x8 zero8() {
  f32x8 z;
#pragma unroll
  for (int i = 0; i < 8; ++i) z[i] = 0.0f;
  return z;
}

DEV f32x8 wmma_bf(bf16x16 a, bf16x16 b, f32x8 c) {
  return __builtin_amdgcn_wmma_f32_16x16x32_bf16(false, a, false, b, (short)0, c,
                                                 false, false);
}

// Load a 16x32 bf16 operand tile (A-layout; also valid for B given row-of-W
// loading) from LDS. Row stride in elements; rows must be 16B aligned.
// Lane l<16 : row l, K = {0..7, 16..23}; lane l>=16: row l-16, K = {8..15, 24..31}.
DEV bf16x16 frag_lds(const bf16* p0, int stride, int lane) {
  const int row = lane & 15, half = lane >> 4;
  const bf16* p = p0 + row * stride + half * 8;
  bf16x8 lo = *(const bf16x8*)(p);
  bf16x8 hi = *(const bf16x8*)(p + 16);
  bf16x16 r;
#pragma unroll
  for (int i = 0; i < 8; ++i) { r[i] = lo[i]; r[i + 8] = hi[i]; }
  return r;
}

// Same operand pattern, loaded directly from global f32 rows (optionally scaled).
DEV bf16x16 frag_global_f32(const float* p0, int stride, int lane, float scale) {
  const int row = lane & 15, half = lane >> 4;
  const float* p = p0 + (size_t)row * stride + half * 8;
  bf16x4 c0 = __builtin_convertvector(*(const f32x4*)(p)      * scale, bf16x4);
  bf16x4 c1 = __builtin_convertvector(*(const f32x4*)(p + 4)  * scale, bf16x4);
  bf16x4 c2 = __builtin_convertvector(*(const f32x4*)(p + 16) * scale, bf16x4);
  bf16x4 c3 = __builtin_convertvector(*(const f32x4*)(p + 20) * scale, bf16x4);
  bf16x16 r;
#pragma unroll
  for (int i = 0; i < 4; ++i) {
    r[i] = c0[i]; r[i + 4] = c1[i]; r[i + 8] = c2[i]; r[i + 12] = c3[i];
  }
  return r;
}

// ---------------------------------------------------------------------------
// QKV projection: for each head h, out[b,h,s,:] = x[b,s,h*64:+64] @ W^T
// grid = (B*S/128, H, 3{q,k,v}), 256 threads (8 waves, 16 rows each).
// ---------------------------------------------------------------------------
__global__ __launch_bounds__(256) void qkv_kernel(
    const float* __restrict__ X, const float* __restrict__ Wq,
    const float* __restrict__ Wk, const float* __restrict__ Wv,
    float* __restrict__ Q, float* __restrict__ K, float* __restrict__ V) {
  __shared__ bf16 Wsm[64][72];
  const int tid = threadIdx.x, lane = tid & 31, w = tid >> 5;
  const int h = blockIdx.y, sel = blockIdx.z;
  const int m0 = blockIdx.x * 128;
  const float* W = (sel == 0) ? Wq : (sel == 1) ? Wk : Wv;
  float* out = (sel == 0) ? Q : (sel == 1) ? K : V;

  for (int i = tid; i < 64 * 16; i += 256) {  // 64x64 f32 -> bf16 LDS
    int row = i >> 4, col = (i & 15) * 4;
    store4bf(&Wsm[row][col], *(const f32x4*)&W[row * 64 + col]);
  }
  __syncthreads();

  const float* arow = X + (size_t)(m0 + w * 16) * Dc + h * 64;
  bf16x16 af[2];
  af[0] = frag_global_f32(arow, Dc, lane, 1.0f);
  af[1] = frag_global_f32(arow + 32, Dc, lane, 1.0f);

  f32x8 acc[4];
#pragma unroll
  for (int nt = 0; nt < 4; ++nt) acc[nt] = zero8();
#pragma unroll
  for (int kk = 0; kk < 2; ++kk)
#pragma unroll
    for (int nt = 0; nt < 4; ++nt)
      acc[nt] = wmma_bf(af[kk], frag_lds(&Wsm[nt * 16][kk * 32], 72, lane), acc[nt]);

  const int half = lane >> 4, cl = lane & 15;
#pragma unroll
  for (int nt = 0; nt < 4; ++nt)
#pragma unroll
    for (int r = 0; r < 8; ++r) {
      int m = m0 + w * 16 + r + half * 8;
      int b = m >> 11, s = m & (Sc - 1);
      out[(((size_t)b * Hc + h) * Sc + s) * 64 + nt * 16 + cl] = acc[nt][r];
    }
}

// ---------------------------------------------------------------------------
// Flash attention: grid = (S/128, B*H). 8 waves, each owns 16 query rows.
// q,k,v are [B,H,S,64]; ctx written as [B,S,D] (head-concatenated).
// ---------------------------------------------------------------------------
__global__ __launch_bounds__(256) void attn_kernel(
    const float* __restrict__ Q, const float* __restrict__ Kg,
    const float* __restrict__ Vg, float* __restrict__ ctx) {
  __shared__ bf16 Ksm[64][72];
  __shared__ bf16 Vsm[64][72];   // transposed: Vsm[d][n]
  __shared__ bf16 Psm[128][72];  // per-wave 16-row slices
  const int tid = threadIdx.x, lane = tid & 31, w = tid >> 5;
  const int half = lane >> 4, cl = lane & 15;
  const int bh = blockIdx.y, b = bh >> 4, h = bh & 15;
  const int s0 = blockIdx.x * 128;
  const size_t base = (size_t)bh * Sc;
  const float scale = 0.125f;  // 1/sqrt(64), folded into q

  const float* qrow = Q + (base + s0 + w * 16) * 64;
  bf16x16 qf[2];
  qf[0] = frag_global_f32(qrow, 64, lane, scale);
  qf[1] = frag_global_f32(qrow + 32, 64, lane, scale);

  f32x8 cacc[4];
#pragma unroll
  for (int dt = 0; dt < 4; ++dt) cacc[dt] = zero8();
  float run_m[8], run_l[8];
#pragma unroll
  for (int r = 0; r < 8; ++r) { run_m[r] = -3.0e38f; run_l[r] = 0.0f; }

  for (int j = 0; j < Sc / 64; ++j) {
    const int kb = j * 64;
    for (int i = tid; i < 64 * 16; i += 256) {  // stage K and V^T
      int row = i >> 4, col = (i & 15) * 4;
      store4bf(&Ksm[row][col], *(const f32x4*)(Kg + (base + kb + row) * 64 + col));
      f32x4 vv = *(const f32x4*)(Vg + (base + kb + row) * 64 + col);
#pragma unroll
      for (int c = 0; c < 4; ++c) Vsm[col + c][row] = f2bf(vv[c]);
    }
    __syncthreads();

    f32x8 sacc[4];
#pragma unroll
    for (int nt = 0; nt < 4; ++nt) sacc[nt] = zero8();
#pragma unroll
    for (int kk = 0; kk < 2; ++kk)
#pragma unroll
      for (int nt = 0; nt < 4; ++nt)
        sacc[nt] = wmma_bf(qf[kk], frag_lds(&Ksm[nt * 16][kk * 32], 72, lane), sacc[nt]);

    // online softmax (rows live in accumulator index r + half*8)
    float fac[8], rs[8];
#pragma unroll
    for (int r = 0; r < 8; ++r) {
      float m = sacc[0][r];
#pragma unroll
      for (int nt = 1; nt < 4; ++nt) m = fmaxf(m, sacc[nt][r]);
#pragma unroll
      for (int o = 1; o < 16; o <<= 1) m = fmaxf(m, __shfl_xor(m, o, 16));
      float nm = fmaxf(run_m[r], m);
      fac[r] = __expf(run_m[r] - nm);
      run_m[r] = nm;
      rs[r] = 0.0f;
    }
#pragma unroll
    for (int nt = 0; nt < 4; ++nt)
#pragma unroll
      for (int r = 0; r < 8; ++r) {
        float p = __expf(sacc[nt][r] - run_m[r]);
        sacc[nt][r] = p;
        rs[r] += p;
      }
#pragma unroll
    for (int r = 0; r < 8; ++r) {
      float s = rs[r];
#pragma unroll
      for (int o = 1; o < 16; o <<= 1) s += __shfl_xor(s, o, 16);
      run_l[r] = run_l[r] * fac[r] + s;
#pragma unroll
      for (int nt = 0; nt < 4; ++nt) cacc[nt][r] *= fac[r];
    }

    // C-layout -> A-layout via wave-private LDS slice
#pragma unroll
    for (int nt = 0; nt < 4; ++nt)
#pragma unroll
      for (int r = 0; r < 8; ++r)
        Psm[w * 16 + r + half * 8][nt * 16 + cl] = f2bf(sacc[nt][r]);

#pragma unroll
    for (int kk = 0; kk < 2; ++kk) {
      bf16x16 pa = frag_lds(&Psm[w * 16][kk * 32], 72, lane);
#pragma unroll
      for (int dt = 0; dt < 4; ++dt)
        cacc[dt] = wmma_bf(pa, frag_lds(&Vsm[dt * 16][kk * 32], 72, lane), cacc[dt]);
    }
    __syncthreads();
  }

#pragma unroll
  for (int dt = 0; dt < 4; ++dt)
#pragma unroll
    for (int r = 0; r < 8; ++r) {
      int s = s0 + w * 16 + r + half * 8;
      ctx[((size_t)b * Sc + s) * Dc + h * 64 + dt * 16 + cl] = cacc[dt][r] / run_l[r];
    }
}

// ---------------------------------------------------------------------------
// Generic GEMM: C[m,n] = sum_k A[m,k]*W[n,k] + bias[n] (+resid[m,n]) (+relu)
// Block tile 128x128x64; 8 waves in a 4x2 grid -> 16 WMMA/wave/stage.
// ---------------------------------------------------------------------------
template <bool RELU>
__global__ __launch_bounds__(256) void gemm_kernel(
    const float* __restrict__ A, int lda, const float* __restrict__ W, int K,
    const float* __restrict__ bias, const float* __restrict__ resid,
    float* __restrict__ C, int ldc) {
  __shared__ bf16 Asm[128][72];
  __shared__ bf16 Bsm[128][72];
  const int tid = threadIdx.x, lane = tid & 31, w = tid >> 5;
  const int wm = (w >> 1) * 32, wn = (w & 1) * 64;
  const int m0 = blockIdx.x * 128, n0 = blockIdx.y * 128;

  f32x8 acc[2][4];
#pragma unroll
  for (int mt = 0; mt < 2; ++mt)
#pragma unroll
    for (int nt = 0; nt < 4; ++nt) acc[mt][nt] = zero8();

  for (int kt = 0; kt < K / 64; ++kt) {
    const int kb = kt * 64;
    for (int i = tid; i < 128 * 16; i += 256) {  // 128x64 f32 each, as f32x4
      int row = i >> 4, col = (i & 15) * 4;
      store4bf(&Asm[row][col], *(const f32x4*)&A[(size_t)(m0 + row) * lda + kb + col]);
      store4bf(&Bsm[row][col], *(const f32x4*)&W[(size_t)(n0 + row) * K + kb + col]);
    }
    __syncthreads();

#pragma unroll
    for (int kk = 0; kk < 2; ++kk) {
      bf16x16 af[2];
#pragma unroll
      for (int mt = 0; mt < 2; ++mt)
        af[mt] = frag_lds(&Asm[wm + mt * 16][kk * 32], 72, lane);
#pragma unroll
      for (int nt = 0; nt < 4; ++nt) {
        bf16x16 bfv = frag_lds(&Bsm[wn + nt * 16][kk * 32], 72, lane);
#pragma unroll
        for (int mt = 0; mt < 2; ++mt) acc[mt][nt] = wmma_bf(af[mt], bfv, acc[mt][nt]);
      }
    }
    __syncthreads();
  }

  const int half = lane >> 4, cl = lane & 15;
#pragma unroll
  for (int nt = 0; nt < 4; ++nt) {
    int col = n0 + wn + nt * 16 + cl;
    float bv = bias ? bias[col] : 0.0f;
#pragma unroll
    for (int mt = 0; mt < 2; ++mt)
#pragma unroll
      for (int r = 0; r < 8; ++r) {
        int row = m0 + wm + mt * 16 + r + half * 8;
        float v = acc[mt][nt][r] + bv;
        if (resid) v += resid[(size_t)row * ldc + col];
        if (RELU) v = fmaxf(v, 0.0f);
        C[(size_t)row * ldc + col] = v;
      }
  }
}

// ---------------------------------------------------------------------------
// In-place LayerNorm over D=1024; one 256-thread block per row.
// ---------------------------------------------------------------------------
__global__ __launch_bounds__(256) void ln_kernel(float* __restrict__ X,
                                                 const float* __restrict__ g,
                                                 const float* __restrict__ bb) {
  __shared__ float red[8];
  const int tid = threadIdx.x, lane = tid & 31, w = tid >> 5;
  float* xr = X + (size_t)blockIdx.x * Dc;
  f32x4 xv = *(const f32x4*)&xr[tid * 4];

  float s = xv[0] + xv[1] + xv[2] + xv[3];
#pragma unroll
  for (int o = 16; o; o >>= 1) s += __shfl_xor(s, o, 32);
  if (lane == 0) red[w] = s;
  __syncthreads();
  float tot = 0;
#pragma unroll
  for (int i = 0; i < 8; ++i) tot += red[i];
  float mean = tot * (1.0f / Dc);
  __syncthreads();

  float d0 = xv[0] - mean, d1 = xv[1] - mean, d2 = xv[2] - mean, d3 = xv[3] - mean;
  float vs = d0 * d0 + d1 * d1 + d2 * d2 + d3 * d3;
#pragma unroll
  for (int o = 16; o; o >>= 1) vs += __shfl_xor(vs, o, 32);
  if (lane == 0) red[w] = vs;
  __syncthreads();
  float vtot = 0;
#pragma unroll
  for (int i = 0; i < 8; ++i) vtot += red[i];
  float rstd = rsqrtf(vtot * (1.0f / Dc) + 1e-5f);

  f32x4 out;
  out[0] = d0 * rstd * g[tid * 4 + 0] + bb[tid * 4 + 0];
  out[1] = d1 * rstd * g[tid * 4 + 1] + bb[tid * 4 + 1];
  out[2] = d2 * rstd * g[tid * 4 + 2] + bb[tid * 4 + 2];
  out[3] = d3 * rstd * g[tid * 4 + 3] + bb[tid * 4 + 3];
  *(f32x4*)&xr[tid * 4] = out;
}

// ---------------------------------------------------------------------------
extern "C" void kernel_launch(void* const* d_in, const int* in_sizes, int n_in,
                              void* d_out, int out_size, void* d_ws, size_t ws_size,
                              hipStream_t stream) {
  const float* x     = (const float*)d_in[0];
  const float* Wq    = (const float*)d_in[1];
  const float* Wk    = (const float*)d_in[2];
  const float* Wv    = (const float*)d_in[3];
  const float* Wo    = (const float*)d_in[4];
  const float* bo    = (const float*)d_in[5];
  const float* ln1g  = (const float*)d_in[6];
  const float* ln1b  = (const float*)d_in[7];
  const float* W1    = (const float*)d_in[8];
  const float* b1    = (const float*)d_in[9];
  const float* W2    = (const float*)d_in[10];
  const float* b2    = (const float*)d_in[11];
  const float* ln2g  = (const float*)d_in[12];
  const float* ln2b  = (const float*)d_in[13];
  float* outf = (float*)d_out;

  float* ws = (float*)d_ws;
  const size_t N1 = (size_t)Bc * Sc * Dc;  // 4,194,304 floats
  float* q    = ws;
  float* kbuf = ws + 1 * N1;
  float* vbuf = ws + 2 * N1;
  float* ctx  = ws + 3 * N1;
  float* y1   = ws + 4 * N1;  // attn-out + resid, then LN1 in-place (= n1)
  float* p0   = ws + 5 * N1;  // layer-output ping
  float* p1   = ws + 6 * N1;  // layer-output pong
  float* hbuf = ws + 7 * N1;  // FFN hidden [B,S,DF]

  const dim3 blk(256);
  for (int l = 0; l < Lc; ++l) {
    const float* IN = (l == 0) ? x : (l == 1) ? p0 : p1;
    float* OUT      = (l == 0) ? p0 : (l == 1) ? p1 : outf;

    qkv_kernel<<<dim3(Bc * Sc / 128, Hc, 3), blk, 0, stream>>>(
        IN, Wq + (size_t)l * Dhc * Dhc, Wk + (size_t)l * Dhc * Dhc,
        Wv + (size_t)l * Dhc * Dhc, q, kbuf, vbuf);

    attn_kernel<<<dim3(Sc / 128, Bc * Hc), blk, 0, stream>>>(q, kbuf, vbuf, ctx);

    gemm_kernel<false><<<dim3(Bc * Sc / 128, Dc / 128), blk, 0, stream>>>(
        ctx, Dc, Wo + (size_t)l * Dc * Dc, Dc, bo + (size_t)l * Dc, IN, y1, Dc);

    ln_kernel<<<Bc * Sc, blk, 0, stream>>>(y1, ln1g + (size_t)l * Dc,
                                           ln1b + (size_t)l * Dc);

    gemm_kernel<true><<<dim3(Bc * Sc / 128, DFc / 128), blk, 0, stream>>>(
        y1, Dc, W1 + (size_t)l * DFc * Dc, Dc, b1 + (size_t)l * DFc, nullptr,
        hbuf, DFc);

    gemm_kernel<false><<<dim3(Bc * Sc / 128, Dc / 128), blk, 0, stream>>>(
        hbuf, DFc, W2 + (size_t)l * Dc * DFc, DFc, b2 + (size_t)l * Dc, y1, OUT, Dc);

    ln_kernel<<<Bc * Sc, blk, 0, stream>>>(OUT, ln2g + (size_t)l * Dc,
                                           ln2b + (size_t)l * Dc);
  }
}
